// WindowAttention_47476568490600
// MI455X (gfx1250) — matrix operand order repaired and checked by reading, not compiled, not run on hardware
//
#include <hip/hip_runtime.h>

#define N_TOK 49
#define NHEAD 16
#define HD    32
#define CDIM  512
#define NWB   1024
#define QK_SCALE 0.17677669529663687f  // 32^-0.5

typedef __attribute__((ext_vector_type(16))) __bf16 v16bf;
typedef __attribute__((ext_vector_type(8)))  __bf16 v8bf;
typedef __attribute__((ext_vector_type(8)))  float  v8f;

// Hardware bf16 conversion (RNE) — let the backend pick v_cvt_*_bf16 ops.
__device__ __forceinline__ __bf16 f2bf(float f) { return (__bf16)f; }

__device__ __forceinline__ v8bf pk8(float4 a, float4 b) {
  v8bf r;
  r[0] = (__bf16)a.x; r[1] = (__bf16)a.y; r[2] = (__bf16)a.z; r[3] = (__bf16)a.w;
  r[4] = (__bf16)b.x; r[5] = (__bf16)b.y; r[6] = (__bf16)b.z; r[7] = (__bf16)b.w;
  return r;
}

union FragU { v16bf v; uint4 q[2]; };

// A fragment (16x32 bf16): lanes 0-15 -> K kbase..+7 / kbase+16..+23,
// lanes 16-31 shifted by 8. Two aligned ds_load_b128.
__device__ __forceinline__ v16bf load_a_frag(const __bf16* base, int pitch,
                                             int mbase, int kbase, int lane) {
  int m  = mbase + (lane & 15);
  int kb = kbase + ((lane < 16) ? 0 : 8);
  const __bf16* p = base + m * pitch + kb;
  FragU f;
  f.q[0] = *(const uint4*)(p);
  f.q[1] = *(const uint4*)(p + 16);
  return f.v;
}

// B fragment (32x16): same K split as A fragment; LDS holds B as [n][k].
__device__ __forceinline__ v16bf load_b_frag(const __bf16* base, int pitch,
                                             int nbase, int kbase, int lane) {
  int n  = nbase + (lane & 15);
  int kb = kbase + ((lane < 16) ? 0 : 8);
  const __bf16* p = base + n * pitch + kb;
  FragU f;
  f.q[0] = *(const uint4*)(p);
  f.q[1] = *(const uint4*)(p + 16);
  return f.v;
}

__device__ __forceinline__ v8f wmma_bf16(v16bf a, v16bf b, v8f c) {
  return __builtin_amdgcn_wmma_f32_16x16x32_bf16(false, a, false, b,
                                                 (short)0, c, false, false);
}

// ---------------------------------------------------------------------------
// Kernel 1: QKV projection. Block = (window b, 64-col tile), 49 rows -> 64.
// Software-pipelined K loop; vectorized fp32->bf16 staging.
// Routes cols to Q (pre-scaled), K, V^T (pitch-64 padded, zero-filled).
// ---------------------------------------------------------------------------
__global__ __launch_bounds__(256) void qkv_kernel(
    const float* __restrict__ x, const float* __restrict__ qkv_w,
    const float* __restrict__ qkv_b,
    __bf16* __restrict__ Qb, __bf16* __restrict__ Kb, __bf16* __restrict__ Vt) {
  __shared__ __align__(16) __bf16 As[64 * 32];
  __shared__ __align__(16) __bf16 Bs[64 * 32];
  const int b = blockIdx.x;
  const int colBase = blockIdx.y * 64;
  const int tid = threadIdx.x, lane = tid & 31, w = tid >> 5;
  const int wm = w >> 1, wn = w & 1;      // 4x2 wave grid: 16 rows x 32 cols each
  const int sr = tid >> 2;                // staging row 0..63
  const int sk = (tid & 3) * 8;           // staging col {0,8,16,24}
  const bool arow_ok = sr < N_TOK;
  const float* aRow = x + (b * N_TOK + sr) * CDIM + sk;
  const float* bRow = qkv_w + (colBase + sr) * CDIM + sk;
  __bf16* aDst = As + sr * 32 + sk;
  __bf16* bDst = Bs + sr * 32 + sk;

  float4 a0, a1, b0, b1;
  auto fetch = [&](int kc) {
    if (arow_ok) {
      a0 = *(const float4*)(aRow + kc);
      a1 = *(const float4*)(aRow + kc + 4);
    } else {
      a0 = make_float4(0.f, 0.f, 0.f, 0.f); a1 = a0;
    }
    b0 = *(const float4*)(bRow + kc);
    b1 = *(const float4*)(bRow + kc + 4);
  };

  v8f acc0 = {}, acc1 = {};
  fetch(0);
  for (int kc = 0; kc < CDIM; kc += 32) {
    __syncthreads();                      // previous chunk's frags consumed
    *(v8bf*)aDst = pk8(a0, a1);
    *(v8bf*)bDst = pk8(b0, b1);
    __syncthreads();
    if (kc + 32 < CDIM) fetch(kc + 32);   // prefetch next chunk (overlaps WMMA)
    v16bf a   = load_a_frag(As, 32, wm * 16, 0, lane);
    v16bf bf0 = load_b_frag(Bs, 32, wn * 32, 0, lane);
    v16bf bf1 = load_b_frag(Bs, 32, wn * 32 + 16, 0, lane);
    acc0 = wmma_bf16(a, bf0, acc0);
    acc1 = wmma_bf16(a, bf1, acc1);
  }

  const int rbase = wm * 16 + ((lane < 16) ? 0 : 8);
  const int cl = lane & 15;
  for (int t = 0; t < 2; ++t) {
    v8f acc = t ? acc1 : acc0;
    int col = colBase + wn * 32 + t * 16 + cl;
#pragma unroll
    for (int r = 0; r < 8; ++r) {
      int row = rbase + r;
      float v = acc[r] + qkv_b[col];
      if (col >= 2 * CDIM) {              // V: store transposed, pitch 64, zero pad
        int c2 = col - 2 * CDIM; int h = c2 >> 5, d = c2 & 31;
        Vt[((b * NHEAD + h) * HD + d) * 64 + row] =
            (row < N_TOK) ? f2bf(v) : f2bf(0.0f);
      } else if (row < N_TOK) {
        if (col < CDIM) {                 // Q, pre-scaled
          int h = col >> 5, d = col & 31;
          Qb[((b * NHEAD + h) * N_TOK + row) * HD + d] = f2bf(v * QK_SCALE);
        } else {                          // K
          int c2 = col - CDIM; int h = c2 >> 5, d = c2 & 31;
          Kb[((b * NHEAD + h) * N_TOK + row) * HD + d] = f2bf(v);
        }
      }
    }
  }
}

// ---------------------------------------------------------------------------
// Kernel 2: fused attention per (window, head): QK^T + bias + mask, softmax,
// P@V. Output AO[b][n][h*32+d] bf16 (A layout for proj GEMM).
// ---------------------------------------------------------------------------
__global__ __launch_bounds__(256) void attn_kernel(
    const __bf16* __restrict__ Qb, const __bf16* __restrict__ Kb,
    const __bf16* __restrict__ Vt, const float* __restrict__ rpb,
    const float* __restrict__ mask, __bf16* __restrict__ AO) {
  __shared__ __align__(16) __bf16 Qs[64 * 32];
  __shared__ __align__(16) __bf16 Ks[64 * 32];
  __shared__ __align__(16) __bf16 Vs[32 * 64];
  __shared__ __align__(16) __bf16 Pb[64 * 64];
  __shared__ float S[64 * 65];
  const int b = blockIdx.x, h = blockIdx.y;
  const int bh = b * NHEAD + h;
  const int tid = threadIdx.x, lane = tid & 31, w = tid >> 5;
  const int wm = w >> 1, wn = w & 1;

  // ---- stage Q, K (64x32, rows>=49 zero) and V^T (32x64, already padded) ----
  {
    const int sr = tid >> 2, sk = (tid & 3) * 8;
    uint4 zq = make_uint4(0u, 0u, 0u, 0u);
    uint4 q = zq, k = zq;
    if (sr < N_TOK) {
      q = *(const uint4*)(Qb + (bh * N_TOK + sr) * HD + sk);
      k = *(const uint4*)(Kb + (bh * N_TOK + sr) * HD + sk);
    }
    *(uint4*)(Qs + sr * 32 + sk) = q;
    *(uint4*)(Ks + sr * 32 + sk) = k;
    const int vd = tid >> 3, vm = (tid & 7) * 8;
    *(uint4*)(Vs + vd * 64 + vm) = *(const uint4*)(Vt + (bh * HD + vd) * 64 + vm);
  }
  __syncthreads();

  // ---- scores = Q K^T (single K chunk, d = 32) + bias + mask ----
  {
    v16bf a   = load_a_frag(Qs, 32, wm * 16, 0, lane);
    v16bf bf0 = load_b_frag(Ks, 32, wn * 32, 0, lane);
    v16bf bf1 = load_b_frag(Ks, 32, wn * 32 + 16, 0, lane);
    v8f c0 = {}, c1 = {};
    c0 = wmma_bf16(a, bf0, c0);
    c1 = wmma_bf16(a, bf1, c1);
    const int rbase = wm * 16 + ((lane < 16) ? 0 : 8);
    const int cl = lane & 15;
    for (int t = 0; t < 2; ++t) {
      v8f c = t ? c1 : c0;
      int col = wn * 32 + t * 16 + cl;
#pragma unroll
      for (int r = 0; r < 8; ++r) {
        int row = rbase + r;
        float sv;
        if (row < N_TOK && col < N_TOK) {
          int idx = ((col / 7) - (row / 7) + 6) * 13;  // rel-pos (row coord only)
          sv = c[r] + rpb[idx * NHEAD + h]
                    + mask[((b & 3) * N_TOK + row) * N_TOK + col];
        } else {
          sv = -1e30f;                    // padded key columns -> zero prob
        }
        S[row * 65 + col] = sv;
      }
    }
  }
  __syncthreads();

  // ---- softmax over padded 64 key columns; one thread per row ----
  if (tid < 64) {
    const float* srow = S + tid * 65;
    float mx = -1e30f;
    for (int m = 0; m < 64; ++m) mx = fmaxf(mx, srow[m]);
    float sum = 0.0f;
    for (int m = 0; m < 64; ++m) sum += __expf(srow[m] - mx);
    float inv = 1.0f / sum;
    unsigned* prow = (unsigned*)(Pb + tid * 64);
    for (int m = 0; m < 64; m += 2) {
      union { __bf16 b[2]; unsigned u; } p;
      p.b[0] = (__bf16)(__expf(srow[m]     - mx) * inv);
      p.b[1] = (__bf16)(__expf(srow[m + 1] - mx) * inv);
      prow[m >> 1] = p.u;
    }
  }
  __syncthreads();

  // ---- out = P @ V  (64x32 output, K = 64 in two chunks) ----
  {
    v8f c = {};
#pragma unroll
    for (int kc = 0; kc < 64; kc += 32) {
      v16bf a  = load_a_frag(Pb, 64, wm * 16, kc, lane);
      v16bf bb = load_b_frag(Vs, 64, wn * 16, kc, lane);
      c = wmma_bf16(a, bb, c);
    }
    const int rbase = wm * 16 + ((lane < 16) ? 0 : 8);
    const int d = wn * 16 + (lane & 15);
#pragma unroll
    for (int r = 0; r < 8; ++r) {
      int row = rbase + r;
      if (row < N_TOK)
        AO[(b * N_TOK + row) * CDIM + h * HD + d] = f2bf(c[r]);
    }
  }
}

// ---------------------------------------------------------------------------
// Kernel 3: output projection, fp32 result + bias. Same pipeline as K1;
// A operand already bf16 in AO.
// ---------------------------------------------------------------------------
__global__ __launch_bounds__(256) void proj_kernel(
    const __bf16* __restrict__ AO, const float* __restrict__ proj_w,
    const float* __restrict__ proj_b, float* __restrict__ out) {
  __shared__ __align__(16) __bf16 As[64 * 32];
  __shared__ __align__(16) __bf16 Bs[64 * 32];
  const int b = blockIdx.x;
  const int colBase = blockIdx.y * 64;
  const int tid = threadIdx.x, lane = tid & 31, w = tid >> 5;
  const int wm = w >> 1, wn = w & 1;
  const int sr = tid >> 2, sk = (tid & 3) * 8;
  const bool arow_ok = sr < N_TOK;
  const __bf16* aRow = AO + (b * N_TOK + sr) * CDIM + sk;
  const float* bRow = proj_w + (colBase + sr) * CDIM + sk;
  __bf16* aDst = As + sr * 32 + sk;
  __bf16* bDst = Bs + sr * 32 + sk;

  uint4 av; float4 b0, b1;
  auto fetch = [&](int kc) {
    av = arow_ok ? *(const uint4*)(aRow + kc) : make_uint4(0u, 0u, 0u, 0u);
    b0 = *(const float4*)(bRow + kc);
    b1 = *(const float4*)(bRow + kc + 4);
  };

  v8f acc0 = {}, acc1 = {};
  fetch(0);
  for (int kc = 0; kc < CDIM; kc += 32) {
    __syncthreads();
    *(uint4*)aDst = av;
    *(v8bf*)bDst = pk8(b0, b1);
    __syncthreads();
    if (kc + 32 < CDIM) fetch(kc + 32);
    v16bf a   = load_a_frag(As, 32, wm * 16, 0, lane);
    v16bf bf0 = load_b_frag(Bs, 32, wn * 32, 0, lane);
    v16bf bf1 = load_b_frag(Bs, 32, wn * 32 + 16, 0, lane);
    acc0 = wmma_bf16(a, bf0, acc0);
    acc1 = wmma_bf16(a, bf1, acc1);
  }

  const int rbase = wm * 16 + ((lane < 16) ? 0 : 8);
  const int cl = lane & 15;
  for (int t = 0; t < 2; ++t) {
    v8f acc = t ? acc1 : acc0;
    int col = colBase + wn * 32 + t * 16 + cl;
#pragma unroll
    for (int r = 0; r < 8; ++r) {
      int row = rbase + r;
      if (row < N_TOK)
        out[(b * N_TOK + row) * CDIM + col] = acc[r] + proj_b[col];
    }
  }
}

extern "C" void kernel_launch(void* const* d_in, const int* in_sizes, int n_in,
                              void* d_out, int out_size, void* d_ws, size_t ws_size,
                              hipStream_t stream) {
  const float* x      = (const float*)d_in[0];
  const float* mask   = (const float*)d_in[1];
  const float* qkv_w  = (const float*)d_in[2];
  const float* qkv_b  = (const float*)d_in[3];
  const float* proj_w = (const float*)d_in[4];
  const float* proj_b = (const float*)d_in[5];
  const float* rpb    = (const float*)d_in[6];
  float* out = (float*)d_out;

  const size_t QK_BYTES = (size_t)NWB * NHEAD * N_TOK * HD * 2;  // 51,380,224
  const size_t VT_BYTES = (size_t)NWB * NHEAD * HD * 64 * 2;     // 67,108,864
  char* ws = (char*)d_ws;
  __bf16* Qb = (__bf16*)(ws);
  __bf16* Kb = (__bf16*)(ws + QK_BYTES);
  __bf16* Vt = (__bf16*)(ws + 2 * QK_BYTES);
  __bf16* AO = (__bf16*)(ws + 2 * QK_BYTES + VT_BYTES);

  qkv_kernel <<<dim3(NWB, 24), 256, 0, stream>>>(x, qkv_w, qkv_b, Qb, Kb, Vt);
  attn_kernel<<<dim3(NWB, NHEAD), 256, 0, stream>>>(Qb, Kb, Vt, rpb, mask, AO);
  proj_kernel<<<dim3(NWB, 8), 256, 0, stream>>>(AO, proj_w, proj_b, out);
}
